// MultiheadAttention_47064251629875
// MI455X (gfx1250) — matrix operand order repaired and checked by reading, not compiled, not run on hardware
//
#include <hip/hip_runtime.h>
#include <hip/hip_bf16.h>

// ---------------------------------------------------------------------------
// MHA forward for MI455X (gfx1250, wave32, WMMA bf16 -> f32 accumulate)
//   x[2,2048,2048] @ Wqkv[2048,3072] + bqkv -> qkv
//   GQA attention (32 q heads, 8 kv heads, hd=64), softmax fp32
//   out @ Wo[2048,2048] + bo -> d_out (fp32)
// GEMMs stage tiles into LDS with GLOBAL_LOAD_ASYNC_TO_LDS_B128 (ASYNCcnt),
// double-buffered.
// ---------------------------------------------------------------------------

typedef __bf16 bf16;
typedef __attribute__((ext_vector_type(4)))  bf16  v4bf;
typedef __attribute__((ext_vector_type(8)))  bf16  v8bf;
typedef __attribute__((ext_vector_type(16))) bf16  v16bf;
typedef __attribute__((ext_vector_type(8)))  float v8f;
typedef int v4i __attribute__((vector_size(16)));

constexpr int BATCH = 2;
constexpr int SEQ   = 2048;
constexpr int DM    = 2048;
constexpr int NQ    = 32;
constexpr int NKV   = 8;
constexpr int HD    = 64;
constexpr int FQKV  = 3072;          // HD*(NQ+2*NKV)
constexpr int MROWS = BATCH * SEQ;   // 4096

#if defined(__AMDGCN__) && __has_builtin(__builtin_amdgcn_global_load_async_to_lds_b128)
#define USE_ASYNC_LDS 1
#else
#define USE_ASYNC_LDS 0
#endif

// Copy 16 bytes global -> LDS. Async (ASYNCcnt) when supported, else via regs.
static __device__ __forceinline__ void cp16_g2l(const bf16* g, bf16* l) {
#if USE_ASYNC_LDS
  typedef __attribute__((address_space(1))) v4i gv4i;
  typedef __attribute__((address_space(3))) v4i lv4i;
  __builtin_amdgcn_global_load_async_to_lds_b128(
      (gv4i*)(uintptr_t)g,
      (lv4i*)(uint32_t)(uintptr_t)l,
      /*offset=*/0, /*cpol=*/0);
#else
  *(v8bf*)l = *(const v8bf*)g;
#endif
}

template <int N>
static __device__ __forceinline__ void wait_async() {
#if USE_ASYNC_LDS
#if __has_builtin(__builtin_amdgcn_s_wait_asynccnt)
  __builtin_amdgcn_s_wait_asynccnt(N);
#else
  asm volatile("s_wait_asynccnt %0" :: "n"(N) : "memory");
#endif
#endif
}

// Build a 16-element bf16 fragment from two contiguous 16-byte chunks.
static __device__ __forceinline__ v16bf frag16(const bf16* lo, const bf16* hi) {
  v8bf a = *(const v8bf*)lo;
  v8bf b = *(const v8bf*)hi;
  v16bf r;
#pragma unroll
  for (int i = 0; i < 8; ++i) { r[i] = a[i]; r[i + 8] = b[i]; }
  return r;
}

static __device__ __forceinline__ v8f wmma_bf(v16bf a, v16bf b, v8f c) {
  // v_wmma_f32_16x16x32_bf16: D = A(16x32) x B(32x16) + C(16x16 f32)
  return __builtin_amdgcn_wmma_f32_16x16x32_bf16(
      /*neg_a=*/false, a, /*neg_b=*/false, b,
      /*c_mod=*/(short)0, c, /*reuse_a=*/false, /*reuse_b=*/false);
}

// ---------------------------------------------------------------------------
// f32 -> bf16 elementwise (vectorized x4)
// ---------------------------------------------------------------------------
__global__ __launch_bounds__(256) void cvt_f32_bf16(const float* __restrict__ src,
                                                    bf16* __restrict__ dst, int n4) {
  int i = blockIdx.x * 256 + threadIdx.x;
  if (i < n4) {
    float4 v = ((const float4*)src)[i];
    v4bf o;
    o[0] = (bf16)v.x; o[1] = (bf16)v.y; o[2] = (bf16)v.z; o[3] = (bf16)v.w;
    ((v4bf*)dst)[i] = o;
  }
}

// ---------------------------------------------------------------------------
// f32 [R][C] -> bf16 [C][R]  (tiled transpose; R,C multiples of 32)
// ---------------------------------------------------------------------------
__global__ __launch_bounds__(256) void transpose_cvt_f32_bf16(
    const float* __restrict__ src, bf16* __restrict__ dst, int R, int C) {
  __shared__ float tile[32][33];
  int c0 = blockIdx.x * 32, r0 = blockIdx.y * 32;
  int tx = threadIdx.x, ty = threadIdx.y;   // block (32,8)
#pragma unroll
  for (int i = 0; i < 32; i += 8)
    tile[ty + i][tx] = src[(size_t)(r0 + ty + i) * C + c0 + tx];
  __syncthreads();
#pragma unroll
  for (int i = 0; i < 32; i += 8)
    dst[(size_t)(c0 + ty + i) * R + r0 + tx] = (bf16)tile[tx][ty + i];
}

// ---------------------------------------------------------------------------
// Extract V from qkv_bf16 and transpose per kv-head: vT[b][kvh][d][s]
// grid: (SEQ/32, HD/32, BATCH*NKV), block (32,8)
// ---------------------------------------------------------------------------
__global__ __launch_bounds__(256) void extract_vT(const bf16* __restrict__ qkv,
                                                  bf16* __restrict__ vT) {
  __shared__ bf16 tile[32][33];
  int z = blockIdx.z;                 // b*8 + kvh
  int b = z >> 3, kvh = z & 7;
  int s0 = blockIdx.x * 32, d0 = blockIdx.y * 32;
  int tx = threadIdx.x, ty = threadIdx.y;
  const bf16* src = qkv + (size_t)b * SEQ * FQKV + (NQ + NKV) * HD + kvh * HD;
#pragma unroll
  for (int i = 0; i < 32; i += 8)
    tile[ty + i][tx] = src[(size_t)(s0 + ty + i) * FQKV + d0 + tx];
  __syncthreads();
#pragma unroll
  for (int i = 0; i < 32; i += 8)
    vT[((size_t)z * HD + d0 + ty + i) * SEQ + s0 + tx] = tile[tx][ty + i];
}

// ---------------------------------------------------------------------------
// GEMM: C[M][N] = A[M][K](bf16) x B (given as BT[N][K] bf16) + bias[N]
// 8 waves/block, block tile 128x128, K-step 32, double-buffered async LDS
// staging. Each wave computes 32x64 (2x4 WMMA subtiles).
// ---------------------------------------------------------------------------
constexpr int LP = 56;  // LDS row pitch (bf16): 112B, 16B-aligned, conflict-free

template <bool OUT_BF16>
__global__ __launch_bounds__(256) void gemm_bf16_wmma(
    const bf16* __restrict__ A, const bf16* __restrict__ BT,
    const float* __restrict__ bias, bf16* __restrict__ Cb,
    float* __restrict__ Cf, int M, int N, int K) {
  __shared__ __align__(16) bf16 sA[2][128 * LP];
  __shared__ __align__(16) bf16 sB[2][128 * LP];

  const int tid   = threadIdx.x;
  const int lane  = tid & 31;
  const int wid   = tid >> 5;                // 0..7
  const int wm    = wid & 3;                 // 4 wave rows
  const int wn    = wid >> 2;                // 2 wave cols
  const int lhalf = lane >> 4;
  const int l15   = lane & 15;
  const int m0    = blockIdx.y * 128;
  const int n0    = blockIdx.x * 128;

  // Stage one 128x32 tile of A and of BT into LDS buffer `buf` at k-offset k.
  // 512 chunks of 16B per tile; 2 chunks of each per thread
  // -> 4 async instructions per wave per stage.
  auto stage = [&](int buf, int k) {
#pragma unroll
    for (int u = 0; u < 2; ++u) {
      const int c   = tid * 2 + u;        // 0..511
      const int row = c >> 2;
      const int cc  = (c & 3) * 8;        // bf16 element offset within row
      cp16_g2l(A  + (size_t)(m0 + row) * K + k + cc, &sA[buf][row * LP + cc]);
      cp16_g2l(BT + (size_t)(n0 + row) * K + k + cc, &sB[buf][row * LP + cc]);
    }
  };

  v8f acc[2][4] = {};
  stage(0, 0);
  const int T = K >> 5;
  for (int i = 0; i < T; ++i) {
    const int buf = i & 1;
    if (i + 1 < T) { stage(buf ^ 1, (i + 1) << 5); wait_async<4>(); }
    else           { wait_async<0>(); }
    __syncthreads();

    const bf16* a = sA[buf];
    const bf16* b = sB[buf];
    const int  ka = lhalf * 8;            // A frag chunks at ka, ka+16
    const bf16* ar0 = a + (wm * 32 + l15) * LP + ka;
    const bf16* ar1 = ar0 + 16 * LP;
    v16bf a0 = frag16(ar0, ar0 + 16);
    v16bf a1 = frag16(ar1, ar1 + 16);
    const int kb = lhalf * 8;             // B frag chunks at kb, kb+16
    v16bf bfr[4];
#pragma unroll
    for (int j = 0; j < 4; ++j) {
      const bf16* br = b + (wn * 64 + j * 16 + l15) * LP + kb;
      bfr[j] = frag16(br, br + 16);
    }
#pragma unroll
    for (int j = 0; j < 4; ++j) {
      acc[0][j] = wmma_bf(a0, bfr[j], acc[0][j]);
      acc[1][j] = wmma_bf(a1, bfr[j], acc[1][j]);
    }
    __syncthreads();
  }

#pragma unroll
  for (int j = 0; j < 4; ++j) {
    const int col = n0 + wn * 64 + j * 16 + l15;
    const float bj = bias[col];
#pragma unroll
    for (int i = 0; i < 2; ++i) {
#pragma unroll
      for (int r = 0; r < 8; ++r) {
        const int row = m0 + wm * 32 + i * 16 + lhalf * 8 + r;
        const float v = acc[i][j][r] + bj;
        if (OUT_BF16) Cb[(size_t)row * N + col] = (bf16)v;
        else          Cf[(size_t)row * N + col] = v;
      }
    }
  }
}

// ---------------------------------------------------------------------------
// Flash-style GQA attention. One wave per (q-tile of 16, head, batch).
// q,k come straight from qkv_bf16 rows (d-contiguous); v from vT.
// ---------------------------------------------------------------------------
__global__ __launch_bounds__(32) void attn_wmma(const bf16* __restrict__ qkv,
                                                const bf16* __restrict__ vT,
                                                bf16* __restrict__ out) {
  __shared__ __align__(16) bf16 Pl[16 * 32];   // P tile (16 q x 32 k) bf16

  const int qb  = blockIdx.x;      // 0..127
  const int h   = blockIdx.y;      // 0..31
  const int b   = blockIdx.z;      // 0..1
  const int kvh = h >> 2;          // n_rep = 4
  const int lane  = threadIdx.x & 31;
  const int lhalf = lane >> 4;
  const int l15   = lane & 15;

  const size_t qkv_b = (size_t)b * SEQ * FQKV;
  const bf16* qrow  = qkv + qkv_b + (size_t)(qb * 16 + l15) * FQKV + h * HD;
  const bf16* kbase = qkv + qkv_b + NQ * HD + kvh * HD;     // + s*FQKV + d
  const bf16* vbase = vT + (size_t)(b * NKV + kvh) * HD * SEQ;

  // Q fragments for the two d-steps (d: 0..31, 32..63)
  v16bf aq[2];
#pragma unroll
  for (int t = 0; t < 2; ++t) {
    const int ka = t * 32 + lhalf * 8;
    aq[t] = frag16(qrow + ka, qrow + ka + 16);
  }

  float m_run[8], l_run[8];
#pragma unroll
  for (int r = 0; r < 8; ++r) { m_run[r] = -1e30f; l_run[r] = 0.0f; }
  v8f o[4] = {};
  const float scale = 0.125f;   // 1/sqrt(64)

  for (int sk = 0; sk < SEQ; sk += 32) {
    // ---- scores: two 16x16 tiles over d=64 (2 WMMA K-steps each) ----
    v8f c[2] = {};
#pragma unroll
    for (int tile = 0; tile < 2; ++tile) {
#pragma unroll
      for (int t = 0; t < 2; ++t) {
        const bf16* kr = kbase + (size_t)(sk + tile * 16 + l15) * FQKV
                                + t * 32 + lhalf * 8;
        v16bf bk = frag16(kr, kr + 16);
        c[tile] = wmma_bf(aq[t], bk, c[tile]);
      }
    }
    // ---- online softmax (row stats via half-wave shuffles) ----
#pragma unroll
    for (int r = 0; r < 8; ++r) {
      float c0 = c[0][r] * scale;
      float c1 = c[1][r] * scale;
      float mx = fmaxf(c0, c1);
#pragma unroll
      for (int off = 1; off < 16; off <<= 1)
        mx = fmaxf(mx, __shfl_xor(mx, off, 32));
      const float mnew = fmaxf(m_run[r], mx);
      const float sc   = __expf(m_run[r] - mnew);
      c0 = __expf(c0 - mnew);
      c1 = __expf(c1 - mnew);
      float srow = c0 + c1;
#pragma unroll
      for (int off = 1; off < 16; off <<= 1)
        srow += __shfl_xor(srow, off, 32);
      l_run[r] = l_run[r] * sc + srow;
      m_run[r] = mnew;
#pragma unroll
      for (int j = 0; j < 4; ++j) o[j][r] *= sc;
      c[0][r] = c0;
      c[1][r] = c1;
    }
    // ---- C-layout -> A-layout for P via LDS ----
    __builtin_amdgcn_wave_barrier();
#pragma unroll
    for (int tile = 0; tile < 2; ++tile)
#pragma unroll
      for (int r = 0; r < 8; ++r)
        Pl[(r + lhalf * 8) * 32 + tile * 16 + l15] = (bf16)c[tile][r];
    asm volatile("s_wait_dscnt 0" ::: "memory");  // LDS RAW hand-off (in-order per wave)
    __builtin_amdgcn_wave_barrier();
    {
      const int ka = lhalf * 8;
      v16bf ap = frag16(&Pl[l15 * 32 + ka], &Pl[l15 * 32 + ka + 16]);
#pragma unroll
      for (int j = 0; j < 4; ++j) {
        const bf16* vr = vbase + (size_t)(j * 16 + l15) * SEQ + sk + lhalf * 8;
        v16bf bv = frag16(vr, vr + 16);
        o[j] = wmma_bf(ap, bv, o[j]);
      }
    }
  }

  // ---- finalize: divide by row sums, store bf16 ----
  bf16* orow = out + ((size_t)b * SEQ + qb * 16) * DM + h * HD;
#pragma unroll
  for (int j = 0; j < 4; ++j) {
#pragma unroll
    for (int r = 0; r < 8; ++r) {
      const float v = o[j][r] / l_run[r];
      orow[(size_t)(r + lhalf * 8) * DM + j * 16 + l15] = (bf16)v;
    }
  }
}

// ---------------------------------------------------------------------------
// Host-side orchestration
// ---------------------------------------------------------------------------
extern "C" void kernel_launch(void* const* d_in, const int* in_sizes, int n_in,
                              void* d_out, int out_size, void* d_ws, size_t ws_size,
                              hipStream_t stream) {
  const float* x    = (const float*)d_in[0];  // [2,2048,2048]
  const float* Wqkv = (const float*)d_in[1];  // [2048,3072]
  const float* bqkv = (const float*)d_in[2];  // [3072]
  const float* Wo   = (const float*)d_in[3];  // [2048,2048]
  const float* bo   = (const float*)d_in[4];  // [2048]
  float* out        = (float*)d_out;          // [2,2048,2048]

  char* ws = (char*)d_ws;
  size_t off = 0;
  auto alloc = [&](size_t bytes) { char* p = ws + off; off += (bytes + 255) & ~size_t(255); return p; };
  bf16* x_bf    = (bf16*)alloc((size_t)MROWS * DM * 2);       // 16.8 MB
  bf16* WqkvT   = (bf16*)alloc((size_t)FQKV * DM * 2);        // 12.6 MB
  bf16* WoT     = (bf16*)alloc((size_t)DM * DM * 2);          //  8.4 MB
  bf16* qkv_bf  = (bf16*)alloc((size_t)MROWS * FQKV * 2);     // 25.2 MB
  bf16* vTw     = (bf16*)alloc((size_t)BATCH * NKV * HD * SEQ * 2); // 4.2 MB
  bf16* attn_o  = (bf16*)alloc((size_t)MROWS * DM * 2);       // 16.8 MB  (~84 MB total)

  // 1) x -> bf16
  {
    int n4 = MROWS * DM / 4;
    cvt_f32_bf16<<<n4 / 256, 256, 0, stream>>>(x, x_bf, n4);
  }
  // 2) Wqkv^T, Wo^T -> bf16
  transpose_cvt_f32_bf16<<<dim3(FQKV / 32, DM / 32), dim3(32, 8), 0, stream>>>(Wqkv, WqkvT, DM, FQKV);
  transpose_cvt_f32_bf16<<<dim3(DM / 32, DM / 32),   dim3(32, 8), 0, stream>>>(Wo, WoT, DM, DM);
  // 3) QKV projection (bf16 out, +bias)
  gemm_bf16_wmma<true><<<dim3(FQKV / 128, MROWS / 128), 256, 0, stream>>>(
      x_bf, WqkvT, bqkv, qkv_bf, nullptr, MROWS, FQKV, DM);
  // 4) v^T extraction
  extract_vT<<<dim3(SEQ / 32, HD / 32, BATCH * NKV), dim3(32, 8), 0, stream>>>(qkv_bf, vTw);
  // 5) attention
  attn_wmma<<<dim3(SEQ / 16, NQ, BATCH), 32, 0, stream>>>(qkv_bf, vTw, attn_o);
  // 6) output projection (fp32 out, +bias)
  gemm_bf16_wmma<false><<<dim3(DM / 128, MROWS / 128), 256, 0, stream>>>(
      attn_o, WoT, bo, nullptr, out, MROWS, DM, DM);
}
